// CausalMultiheadSelfAttention_86852828660387
// MI455X (gfx1250) — matrix operand order repaired and checked by reading, not compiled, not run on hardware
//
#include <hip/hip_runtime.h>

typedef __attribute__((ext_vector_type(16))) __bf16 v16bf;
typedef __attribute__((ext_vector_type(8))) float v8f;
typedef unsigned short u16;
typedef unsigned int u32;

#define D_MODEL 1024
#define NHEADS 16
#define DK 64
#define SEQ 2048
#define BATCH 4

__device__ __forceinline__ u16 f2bf(float f) {
    u32 u = __float_as_uint(f);
    u32 r = u + 0x7FFFu + ((u >> 16) & 1u);
    return (u16)(r >> 16);
}

__device__ __forceinline__ v8f v8f_zero() {
    v8f z = {0.f, 0.f, 0.f, 0.f, 0.f, 0.f, 0.f, 0.f};
    return z;
}

union FragB { v16bf v; u32 u[8]; uint2 q[4]; };

// D = A*B + C, bf16 in / f32 accumulate, 16x16x32
#define WMMA_BF16(a, b, c) \
    __builtin_amdgcn_wmma_f32_16x16x32_bf16(false, (a), false, (b), (short)0, (c), false, false)

// A fragment (16 rows x 32 K) from a bf16 row-major tile.
// row = lane&15; VGPR0..3 hold K=(half?8:0)+0..7, VGPR4..7 hold K=16+(half?8:0)+0..7
__device__ __forceinline__ void load_a(FragB& f, const u16* tile, int pitch, int lane) {
    int r0 = lane & 15, hh = (lane >> 4) & 1;
    const u16* p = tile + r0 * pitch + hh * 8;
    f.q[0] = *(const uint2*)(p);
    f.q[1] = *(const uint2*)(p + 4);
    f.q[2] = *(const uint2*)(p + 16);
    f.q[3] = *(const uint2*)(p + 20);
}

// B fragment (32 K x 16 N) from an N-major tile (tile[n][k], k contiguous).
// n = lane&15; VGPR0..3 hold K=(half?8:0)+0..7, VGPR4..7 hold K=16+(half?8:0)+0..7
__device__ __forceinline__ void load_b(FragB& f, const u16* tile, int pitch, int lane) {
    int r0 = lane & 15, hh = (lane >> 4) & 1;
    const u16* p = tile + r0 * pitch + hh * 8;
    f.q[0] = *(const uint2*)(p);
    f.q[1] = *(const uint2*)(p + 4);
    f.q[2] = *(const uint2*)(p + 16);
    f.q[3] = *(const uint2*)(p + 20);
}

// ---------------------------------------------------------------------------
// fp32 -> bf16 conversion
// ---------------------------------------------------------------------------
__global__ void f32_to_bf16_kernel(const float* __restrict__ in, u16* __restrict__ out, int n) {
    int i = blockIdx.x * blockDim.x + threadIdx.x;
    int stride = gridDim.x * blockDim.x;
    for (; i < n; i += stride) out[i] = f2bf(in[i]);
}

// ---------------------------------------------------------------------------
// Projection GEMM: out = X @ W^T, X:[Nrows,1024] bf16, W:[1024 out,1024 in] bf16
// mode 0: bf16 out, split-head [B,H,S,dk]
// mode 1: fp32 out, [Nrows,1024]
// mode 2: bf16 out, transposed per head [B,H,dk,S]   (V)
// Double-buffered LDS: one barrier per k-chunk.
// ---------------------------------------------------------------------------
#define PPITCH 40  // 32 + 8 bf16 padding
#define NKC (D_MODEL / 32)

__global__ __launch_bounds__(256) void proj_gemm(
    const u16* __restrict__ X, const u16* __restrict__ W,
    void* __restrict__ outp, int mode, float scale)
{
    __shared__ u16 lA[2][128 * PPITCH];
    __shared__ u16 lB[2][128 * PPITCH];

    const int tid = threadIdx.x;
    const int lane = tid & 31;
    const int wave = tid >> 5;
    const int wm = wave >> 1;    // 0..3 : 32-row strip
    const int wn = wave & 1;     // 0..1 : 64-col strip
    const int rowBase = blockIdx.x * 128;
    const int colBase = blockIdx.y * 128;

    const int ldRow = tid >> 1;        // 0..127
    const int ldSeg = (tid & 1) * 16;  // 0 or 16 elems

    const u16* gxBase = X + (long)(rowBase + ldRow) * D_MODEL + ldSeg;
    const u16* gwBase = W + (long)(colBase + ldRow) * D_MODEL + ldSeg;
    u16* const daBase = &lA[0][0] + ldRow * PPITCH + ldSeg;
    u16* const dbBase = &lB[0][0] + ldRow * PPITCH + ldSeg;
    const int bufStride = 128 * PPITCH;

    v8f acc[2][4];
#pragma unroll
    for (int i = 0; i < 2; ++i)
#pragma unroll
        for (int j = 0; j < 4; ++j) acc[i][j] = v8f_zero();

    // stage chunk 0 into buffer 0
    {
        const u16* gx = gxBase;
        uint4 a0 = *(const uint4*)(gx);
        uint4 a1 = *(const uint4*)(gx + 8);
        *(uint2*)(daBase + 0)  = make_uint2(a0.x, a0.y);
        *(uint2*)(daBase + 4)  = make_uint2(a0.z, a0.w);
        *(uint2*)(daBase + 8)  = make_uint2(a1.x, a1.y);
        *(uint2*)(daBase + 12) = make_uint2(a1.z, a1.w);
        const u16* gw = gwBase;
        uint4 b0 = *(const uint4*)(gw);
        uint4 b1 = *(const uint4*)(gw + 8);
        *(uint2*)(dbBase + 0)  = make_uint2(b0.x, b0.y);
        *(uint2*)(dbBase + 4)  = make_uint2(b0.z, b0.w);
        *(uint2*)(dbBase + 8)  = make_uint2(b1.x, b1.y);
        *(uint2*)(dbBase + 12) = make_uint2(b1.z, b1.w);
    }
    __syncthreads();

    for (int kc = 0; kc < NKC; ++kc) {
        const int cur = kc & 1;
        const int nxt = cur ^ 1;

        // stage next chunk into the other buffer (overlaps with compute below)
        if (kc + 1 < NKC) {
            const u16* gx = gxBase + (kc + 1) * 32;
            uint4 a0 = *(const uint4*)(gx);
            uint4 a1 = *(const uint4*)(gx + 8);
            u16* da = daBase + nxt * bufStride;
            *(uint2*)(da + 0)  = make_uint2(a0.x, a0.y);
            *(uint2*)(da + 4)  = make_uint2(a0.z, a0.w);
            *(uint2*)(da + 8)  = make_uint2(a1.x, a1.y);
            *(uint2*)(da + 12) = make_uint2(a1.z, a1.w);
            const u16* gw = gwBase + (kc + 1) * 32;
            uint4 b0 = *(const uint4*)(gw);
            uint4 b1 = *(const uint4*)(gw + 8);
            u16* db = dbBase + nxt * bufStride;
            *(uint2*)(db + 0)  = make_uint2(b0.x, b0.y);
            *(uint2*)(db + 4)  = make_uint2(b0.z, b0.w);
            *(uint2*)(db + 8)  = make_uint2(b1.x, b1.y);
            *(uint2*)(db + 12) = make_uint2(b1.z, b1.w);
            if (kc + 2 < NKC) {  // hint GL2 for the chunk after that
                __builtin_prefetch(gxBase + (kc + 2) * 32, 0, 1);
                __builtin_prefetch(gwBase + (kc + 2) * 32, 0, 1);
            }
        }

        FragB af[2], bfr[4];
#pragma unroll
        for (int i = 0; i < 2; ++i)
            load_a(af[i], &lA[cur][0] + (wm * 32 + i * 16) * PPITCH, PPITCH, lane);
#pragma unroll
        for (int j = 0; j < 4; ++j)
            load_b(bfr[j], &lB[cur][0] + (wn * 64 + j * 16) * PPITCH, PPITCH, lane);
#pragma unroll
        for (int i = 0; i < 2; ++i)
#pragma unroll
            for (int j = 0; j < 4; ++j)
                acc[i][j] = WMMA_BF16(af[i].v, bfr[j].v, acc[i][j]);

        __syncthreads();
    }

    // C/D layout: n = lane&15 ; m = vgpr + (half? 8 : 0)
    const int r0 = lane & 15, hh = (lane >> 4) & 1;
#pragma unroll
    for (int i = 0; i < 2; ++i) {
#pragma unroll
        for (int j = 0; j < 4; ++j) {
#pragma unroll
            for (int r = 0; r < 8; ++r) {
                int m = rowBase + wm * 32 + i * 16 + r + hh * 8;
                int n = colBase + wn * 64 + j * 16 + r0;
                float v = acc[i][j][r] * scale;
                if (mode == 1) {
                    ((float*)outp)[(long)m * D_MODEL + n] = v;
                } else {
                    int b = m >> 11, s = m & (SEQ - 1);
                    int h = n >> 6, d = n & (DK - 1);
                    long idx;
                    if (mode == 0)
                        idx = ((long)(b * NHEADS + h) * SEQ + s) * DK + d;
                    else
                        idx = ((long)(b * NHEADS + h) * DK + d) * SEQ + s;
                    ((u16*)outp)[idx] = f2bf(v);
                }
            }
        }
    }
}

// ---------------------------------------------------------------------------
// Flash attention: Q [B,H,S,dk] (pre-scaled), K [B,H,S,dk], Vt [B,H,dk,S], bf16
// O [B,S,D] bf16. One workgroup per (b,h,128-query block); wave owns 16 queries.
// Double-buffered K/V tiles: one barrier per key block.
// ---------------------------------------------------------------------------
#define KPITCH 72   // 64 + 8
#define VPITCH 40   // 32 + 8
#define PPITCH2 40  // 32 + 8

__global__ __launch_bounds__(256) void attn_kernel(
    const u16* __restrict__ Q, const u16* __restrict__ K,
    const u16* __restrict__ Vt, u16* __restrict__ O)
{
    __shared__ u16 lK[2][32 * KPITCH];
    __shared__ u16 lV[2][64 * VPITCH];
    __shared__ u16 lP[8 * 16 * PPITCH2];

    const int tid = threadIdx.x;
    const int lane = tid & 31;
    const int wave = tid >> 5;
    const int r0 = lane & 15;
    const int hh = (lane >> 4) & 1;

    const int qb = blockIdx.x & (SEQ / 128 - 1);  // 16 query blocks
    const int bh = blockIdx.x >> 4;               // b*16 + h
    const int q0 = qb * 128 + wave * 16;

    // staging thread mapping
    const int kRow = tid >> 3, kSeg = (tid & 7) * 8;   // K tile: 32 x 64
    const int vRow = tid >> 2, vSeg = (tid & 3) * 8;   // V tile: 64 x 32
    const u16* gK = K + ((long)bh * SEQ + kRow) * DK + kSeg;
    const u16* gV = Vt + ((long)bh * DK + vRow) * SEQ + vSeg;
    u16* const dK = &lK[0][0] + kRow * KPITCH + kSeg;
    u16* const dV = &lV[0][0] + vRow * VPITCH + vSeg;

    // Resident Q fragments: 16 queries x dk(64) = 2 A-fragments
    FragB qf[2];
    {
        const u16* qrow = Q + ((long)bh * SEQ + q0 + r0) * DK;
#pragma unroll
        for (int c = 0; c < 2; ++c) {
            const u16* p = qrow + c * 32 + hh * 8;
            qf[c].q[0] = *(const uint2*)(p);
            qf[c].q[1] = *(const uint2*)(p + 4);
            qf[c].q[2] = *(const uint2*)(p + 16);
            qf[c].q[3] = *(const uint2*)(p + 20);
        }
    }

    float mrow[8], lsum[8];
#pragma unroll
    for (int r = 0; r < 8; ++r) { mrow[r] = -3.0e38f; lsum[r] = 0.0f; }
    v8f oa[4];
#pragma unroll
    for (int t = 0; t < 4; ++t) oa[t] = v8f_zero();

    const int kbMax = qb * 4 + 4;  // causal bound for the whole 128-query block

    // stage key block 0 into buffer 0
    {
        uint4 v = *(const uint4*)gK;
        *(uint2*)(dK + 0) = make_uint2(v.x, v.y);
        *(uint2*)(dK + 4) = make_uint2(v.z, v.w);
        uint4 w = *(const uint4*)gV;
        *(uint2*)(dV + 0) = make_uint2(w.x, w.y);
        *(uint2*)(dV + 4) = make_uint2(w.z, w.w);
    }
    __syncthreads();

    for (int kb = 0; kb < kbMax; ++kb) {
        const int cur = kb & 1;
        const int nxt = cur ^ 1;

        if (kb + 1 < kbMax) {  // stage next block (overlaps with compute)
            uint4 v = *(const uint4*)(gK + (long)(kb + 1) * 32 * DK);
            u16* d = dK + nxt * (32 * KPITCH);
            *(uint2*)(d + 0) = make_uint2(v.x, v.y);
            *(uint2*)(d + 4) = make_uint2(v.z, v.w);
            uint4 w = *(const uint4*)(gV + (kb + 1) * 32);
            u16* e = dV + nxt * (64 * VPITCH);
            *(uint2*)(e + 0) = make_uint2(w.x, w.y);
            *(uint2*)(e + 4) = make_uint2(w.z, w.w);
        }

        // wave-uniform skip: keeps EXEC all-ones inside (WMMA requirement)
        if (kb * 32 <= q0 + 15) {
            // ---- scores: S = Q K^T, two 16-key tiles ----
            v8f st[2];
#pragma unroll
            for (int nt = 0; nt < 2; ++nt) {
                v8f a = v8f_zero();
#pragma unroll
                for (int c = 0; c < 2; ++c) {
                    FragB kf;
                    const u16* p = &lK[cur][0] + (nt * 16 + r0) * KPITCH + c * 32 + hh * 8;
                    kf.q[0] = *(const uint2*)(p);
                    kf.q[1] = *(const uint2*)(p + 4);
                    kf.q[2] = *(const uint2*)(p + 16);
                    kf.q[3] = *(const uint2*)(p + 20);
                    a = WMMA_BF16(qf[c].v, kf.v, a);
                }
                st[nt] = a;
            }

            // ---- causal mask + online softmax ----
            int key0 = kb * 32 + r0;
            int key1 = key0 + 16;
#pragma unroll
            for (int r = 0; r < 8; ++r) {
                int q = q0 + r + hh * 8;
                float s0 = (key0 <= q) ? st[0][r] : -3.0e38f;
                float s1 = (key1 <= q) ? st[1][r] : -3.0e38f;
                float x = fmaxf(s0, s1);
                x = fmaxf(x, __shfl_xor(x, 1, 32));
                x = fmaxf(x, __shfl_xor(x, 2, 32));
                x = fmaxf(x, __shfl_xor(x, 4, 32));
                x = fmaxf(x, __shfl_xor(x, 8, 32));
                float nm = fmaxf(mrow[r], x);
                float alpha = __expf(mrow[r] - nm);
                mrow[r] = nm;
                float p0 = __expf(s0 - nm);
                float p1 = __expf(s1 - nm);
                float ps = p0 + p1;
                ps += __shfl_xor(ps, 1, 32);
                ps += __shfl_xor(ps, 2, 32);
                ps += __shfl_xor(ps, 4, 32);
                ps += __shfl_xor(ps, 8, 32);
                lsum[r] = lsum[r] * alpha + ps;
#pragma unroll
                for (int t = 0; t < 4; ++t) oa[t][r] *= alpha;
                // C-layout -> LDS (wave-private; same-wave DS ops are in order)
                u16* pb = lP + (wave * 16 + r + hh * 8) * PPITCH2;
                pb[r0]      = f2bf(p0);
                pb[16 + r0] = f2bf(p1);
            }

            // ---- O += P V ----
            FragB pf;
            {
                const u16* p = lP + (wave * 16 + r0) * PPITCH2 + hh * 8;
                pf.q[0] = *(const uint2*)(p);
                pf.q[1] = *(const uint2*)(p + 4);
                pf.q[2] = *(const uint2*)(p + 16);
                pf.q[3] = *(const uint2*)(p + 20);
            }
#pragma unroll
            for (int t = 0; t < 4; ++t) {
                FragB vf;
                const u16* p = &lV[cur][0] + (t * 16 + r0) * VPITCH + hh * 8;
                vf.q[0] = *(const uint2*)(p);
                vf.q[1] = *(const uint2*)(p + 4);
                vf.q[2] = *(const uint2*)(p + 16);
                vf.q[3] = *(const uint2*)(p + 20);
                oa[t] = WMMA_BF16(pf.v, vf.v, oa[t]);
            }
        }

        __syncthreads();
    }

    // ---- normalize + write O [B,S,D] bf16 ----
    const int b = bh >> 4, h = bh & 15;
#pragma unroll
    for (int r = 0; r < 8; ++r) {
        float inv = 1.0f / lsum[r];
        int q = q0 + r + hh * 8;
        u16* orow = O + ((long)b * SEQ + q) * D_MODEL + h * DK;
#pragma unroll
        for (int t = 0; t < 4; ++t)
            orow[t * 16 + r0] = f2bf(oa[t][r] * inv);
    }
}

// ---------------------------------------------------------------------------
extern "C" void kernel_launch(void* const* d_in, const int* in_sizes, int n_in,
                              void* d_out, int out_size, void* d_ws, size_t ws_size,
                              hipStream_t stream) {
    const float* x  = (const float*)d_in[0];
    const float* Wq = (const float*)d_in[1];
    const float* Wk = (const float*)d_in[2];
    const float* Wv = (const float*)d_in[3];
    const float* Wo = (const float*)d_in[4];

    char* ws = (char*)d_ws;
    const size_t MB = 1024 * 1024;
    u16* xb  = (u16*)(ws + 0);        // 16 MB
    u16* wqb = (u16*)(ws + 16 * MB);  //  2 MB
    u16* wkb = (u16*)(ws + 18 * MB);
    u16* wvb = (u16*)(ws + 20 * MB);
    u16* wob = (u16*)(ws + 22 * MB);
    u16* Qb  = (u16*)(ws + 24 * MB);  // 16 MB, [B,H,S,dk], pre-scaled by 1/8
    u16* Kb  = (u16*)(ws + 40 * MB);  // 16 MB, [B,H,S,dk]
    u16* Vtb = (u16*)(ws + 56 * MB);  // 16 MB, [B,H,dk,S]
    u16* Ob  = (u16*)(ws + 72 * MB);  // 16 MB, [B,S,D]

    const int nx = BATCH * SEQ * D_MODEL;
    const int nw = D_MODEL * D_MODEL;
    f32_to_bf16_kernel<<<1024, 256, 0, stream>>>(x, xb, nx);
    f32_to_bf16_kernel<<<256, 256, 0, stream>>>(Wq, wqb, nw);
    f32_to_bf16_kernel<<<256, 256, 0, stream>>>(Wk, wkb, nw);
    f32_to_bf16_kernel<<<256, 256, 0, stream>>>(Wv, wvb, nw);
    f32_to_bf16_kernel<<<256, 256, 0, stream>>>(Wo, wob, nw);

    dim3 pg(BATCH * SEQ / 128, D_MODEL / 128);
    proj_gemm<<<pg, 256, 0, stream>>>(xb, wqb, Qb, 0, 0.125f);  // Q, scaled 1/sqrt(dk)
    proj_gemm<<<pg, 256, 0, stream>>>(xb, wkb, Kb, 0, 1.0f);    // K
    proj_gemm<<<pg, 256, 0, stream>>>(xb, wvb, Vtb, 2, 1.0f);   // V (transposed)

    attn_kernel<<<BATCH * NHEADS * (SEQ / 128), 256, 0, stream>>>(Qb, Kb, Vtb, Ob);

    proj_gemm<<<pg, 256, 0, stream>>>(Ob, wob, d_out, 1, 1.0f); // O @ Wo^T -> fp32
}
